// SpectralNormalizedLinear_32409823216153
// MI455X (gfx1250) — compile-verified, matrix operand order checked by reading, not run on hardware
//
#include <hip/hip_runtime.h>
#include <hip/hip_bf16.h>

// ---------------------------------------------------------------------------
// Types for CDNA5 WMMA
// ---------------------------------------------------------------------------
typedef __attribute__((ext_vector_type(16))) __bf16          v16bf;
typedef __attribute__((ext_vector_type(8)))  float           v8f;
typedef __attribute__((ext_vector_type(16))) unsigned short  v16us;
typedef __attribute__((ext_vector_type(8)))  unsigned short  u16x8;

// ---------------------------------------------------------------------------
// bf16 split helpers (round-to-nearest-even), pure bit ops
// ---------------------------------------------------------------------------
__device__ __forceinline__ unsigned short f2bf(float f) {
    unsigned int u = __builtin_bit_cast(unsigned int, f);
    u += 0x7FFFu + ((u >> 16) & 1u);
    return (unsigned short)(u >> 16);
}
__device__ __forceinline__ float bf2f(unsigned short h) {
    unsigned int u = ((unsigned int)h) << 16;
    return __builtin_bit_cast(float, u);
}
__device__ __forceinline__ void split2(float f, unsigned short& hi, unsigned short& lo) {
    hi = f2bf(f);
    lo = f2bf(f - bf2f(hi));
}

// Load a 16-half A/B fragment from an LDS row laid out K-major.
// ISA 16-bit operand layout: lanes 0-15 use K = {kb..kb+7, kb+16..kb+23} with
// kb = 0; lanes 16-31 use kb = 8. Two contiguous 16B loads per lane.
__device__ __forceinline__ v16bf ldfrag(const unsigned short* rowp, int kbase) {
    u16x8 a = *(const u16x8*)(rowp + kbase);
    u16x8 b = *(const u16x8*)(rowp + kbase + 16);
    v16us v = __builtin_shufflevector(a, b, 0, 1, 2, 3, 4, 5, 6, 7,
                                            8, 9, 10, 11, 12, 13, 14, 15);
    return __builtin_bit_cast(v16bf, v);
}

// ---------------------------------------------------------------------------
// Kernel 1: zero the scalar/atomic slot area, compute ||u0||^2 -> scal[0]
// ---------------------------------------------------------------------------
__global__ void init_norm_kernel(const float* __restrict__ u0,
                                 float* __restrict__ scal, int n) {
    __shared__ float red[256];
    int t = threadIdx.x;
    scal[t] = 0.0f;                       // zero all 256 scalar slots
    float s = 0.0f;
    for (int j = t; j < n; j += 256) { float v = u0[j]; s += v * v; }
    red[t] = s;
    __syncthreads();
    for (int off = 128; off > 0; off >>= 1) {
        if (t < off) red[t] += red[t + off];
        __syncthreads();
    }
    if (t == 0) scal[0] = red[0];
}

// ---------------------------------------------------------------------------
// Kernel 2: C = M^T * M  (n x n, f32).  Used for A = W^T W and, since A is
// symmetric, A2 = A^T A = A*A.  64x64 block tile, 256 threads, 4x4 per thread.
// ---------------------------------------------------------------------------
__global__ __launch_bounds__(256)
void mtm_kernel(const float* __restrict__ M, float* __restrict__ C, int n) {
    __shared__ float Ta[16][66];
    __shared__ float Tb[16][66];
    const int tid = threadIdx.x;
    const int tx = tid & 15;
    const int ty = tid >> 4;
    const int i0 = blockIdx.y * 64;
    const int j0 = blockIdx.x * 64;
    float c[4][4] = {};
    for (int k0 = 0; k0 < n; k0 += 16) {
#pragma unroll
        for (int t = 0; t < 4; ++t) {
            int idx = tid + t * 256;          // 0..1023
            int kk = idx >> 6;
            int cc = idx & 63;
            Ta[kk][cc] = M[(size_t)(k0 + kk) * n + i0 + cc];
            Tb[kk][cc] = M[(size_t)(k0 + kk) * n + j0 + cc];
        }
        __syncthreads();
#pragma unroll
        for (int kk = 0; kk < 16; ++kk) {
            float a[4], b[4];
#pragma unroll
            for (int r = 0; r < 4; ++r) a[r] = Ta[kk][ty * 4 + r];
#pragma unroll
            for (int q = 0; q < 4; ++q) b[q] = Tb[kk][tx * 4 + q];
#pragma unroll
            for (int r = 0; r < 4; ++r)
#pragma unroll
                for (int q = 0; q < 4; ++q) c[r][q] = fmaf(a[r], b[q], c[r][q]);
        }
        __syncthreads();
    }
#pragma unroll
    for (int r = 0; r < 4; ++r)
#pragma unroll
        for (int q = 0; q < 4; ++q)
            C[(size_t)(i0 + ty * 4 + r) * n + j0 + tx * 4 + q] = c[r][q];
}

// ---------------------------------------------------------------------------
// Kernel 3: y = M * (x * rsqrt(norm_in + eps)); norm_out += ||y||^2 (atomic).
// Block-per-row; M rows are contiguous (A is symmetric, so u*A == A*u, and
// v = u W^T is exactly row-dots of W).
// ---------------------------------------------------------------------------
__global__ __launch_bounds__(256)
void gemv_norm_kernel(const float* __restrict__ M, const float* __restrict__ x,
                      float* __restrict__ y, const float* __restrict__ nin,
                      float* __restrict__ nout, int n) {
    __shared__ float red[256];
    const int row = blockIdx.x;
    const int t = threadIdx.x;
    const float rs = rsqrtf(nin[0] + 1e-6f);
    const float* Mr = M + (size_t)row * n;
    float s = 0.0f;
    for (int j = t; j < n; j += 256) s = fmaf(Mr[j], x[j], s);
    red[t] = s;
    __syncthreads();
    for (int off = 128; off > 0; off >>= 1) {
        if (t < off) red[t] += red[t + off];
        __syncthreads();
    }
    if (t == 0) {
        float r = red[0] * rs;
        y[row] = r;
        atomicAdd(nout, r * r);
    }
}

// ---------------------------------------------------------------------------
// Kernel 4: sigma = ||u99 A|| / ||u99 W^T||; inv = 1/max(1, sigma/0.9)
// ---------------------------------------------------------------------------
__global__ void factor_kernel(float* __restrict__ scal) {
    float sigma = sqrtf(scal[61]) * rsqrtf(scal[60] + 1e-6f);
    float f = fmaxf(1.0f, sigma * (1.0f / 0.9f));
    scal[64] = 1.0f / f;
}

// ---------------------------------------------------------------------------
// Kernel 5: out = x @ (W * inv) + b  via bf16x3 WMMA (hi*hi + hi*lo + lo*hi).
// 128x128 block tile, 8 wave32 waves, each wave = 32x64 (2x4 WMMA tiles),
// K-step 32.  W chunk stored K-transposed in LDS so both fragments are
// contiguous 16B loads; rows padded to 40 ushorts (80B) to spread LDS banks.
// ---------------------------------------------------------------------------
__global__ __launch_bounds__(256)
void gemm_bf16x3_kernel(const float* __restrict__ X, const float* __restrict__ W,
                        const float* __restrict__ Bv, const float* __restrict__ invp,
                        float* __restrict__ O, int Mdim, int Ndim, int Kdim) {
    __shared__ unsigned short sXh[128][40];
    __shared__ unsigned short sXl[128][40];
    __shared__ unsigned short sWh[128][40];   // [col][k]
    __shared__ unsigned short sWl[128][40];

    const int tid = threadIdx.x;
    const int lane = tid & 31;
    const int wv = tid >> 5;          // 0..7
    const int wr = wv >> 1;           // 0..3  (row group of 32)
    const int wc = wv & 1;            // 0..1  (col group of 64)
    const int lrow = lane & 15;
    const int khalf = lane >> 4;      // 0/1 -> K-base 0/8
    const int kbase = khalf * 8;
    const int m0 = blockIdx.y * 128;
    const int n0 = blockIdx.x * 128;
    const float inv = invp[0];

    v8f acc[2][4] = {};

    for (int kt = 0; kt < Kdim; kt += 32) {
        // ---- stage X tile [128 rows x 32 k] as bf16 hi/lo ----
#pragma unroll
        for (int i = 0; i < 4; ++i) {
            int idx4 = tid + i * 256;                 // 0..1023 float4 slots
            int row = idx4 >> 3;                      // 8 float4 per row
            int kq = (idx4 & 7) << 2;
            float4 v = *(const float4*)(X + (size_t)(m0 + row) * Kdim + kt + kq);
            float f[4] = {v.x, v.y, v.z, v.w};
#pragma unroll
            for (int c = 0; c < 4; ++c) {
                unsigned short hi, lo;
                split2(f[c], hi, lo);
                sXh[row][kq + c] = hi;
                sXl[row][kq + c] = lo;
            }
        }
        // ---- stage W tile [32 k x 128 cols] transposed, scaled by inv ----
#pragma unroll
        for (int i = 0; i < 4; ++i) {
            int idx4 = tid + i * 256;
            int kr = idx4 >> 5;                       // 32 float4 per k-row
            int cq = (idx4 & 31) << 2;
            float4 v = *(const float4*)(W + (size_t)(kt + kr) * Ndim + n0 + cq);
            float f[4] = {v.x * inv, v.y * inv, v.z * inv, v.w * inv};
#pragma unroll
            for (int c = 0; c < 4; ++c) {
                unsigned short hi, lo;
                split2(f[c], hi, lo);
                sWh[cq + c][kr] = hi;
                sWl[cq + c][kr] = lo;
            }
        }
        __syncthreads();

        // ---- fragments ----
        v16bf axh[2], axl[2], bwh[4], bwl[4];
#pragma unroll
        for (int tr = 0; tr < 2; ++tr) {
            int r = wr * 32 + tr * 16 + lrow;
            axh[tr] = ldfrag(&sXh[r][0], kbase);
            axl[tr] = ldfrag(&sXl[r][0], kbase);
        }
#pragma unroll
        for (int tc = 0; tc < 4; ++tc) {
            int c = wc * 64 + tc * 16 + lrow;
            bwh[tc] = ldfrag(&sWh[c][0], kbase);
            bwl[tc] = ldfrag(&sWl[c][0], kbase);
        }

        // ---- 2x4 tiles, 3 WMMA each (bf16x3) ----
#pragma unroll
        for (int tr = 0; tr < 2; ++tr) {
#pragma unroll
            for (int tc = 0; tc < 4; ++tc) {
                acc[tr][tc] = __builtin_amdgcn_wmma_f32_16x16x32_bf16(
                    false, axh[tr], false, bwh[tc], (short)0, acc[tr][tc], false, false);
                acc[tr][tc] = __builtin_amdgcn_wmma_f32_16x16x32_bf16(
                    false, axh[tr], false, bwl[tc], (short)0, acc[tr][tc], false, false);
                acc[tr][tc] = __builtin_amdgcn_wmma_f32_16x16x32_bf16(
                    false, axl[tr], false, bwh[tc], (short)0, acc[tr][tc], false, false);
            }
        }
        __syncthreads();
    }

    // ---- epilogue: C/D layout VGPR r -> M = r + 8*khalf, N = lane%16 ----
#pragma unroll
    for (int tr = 0; tr < 2; ++tr) {
#pragma unroll
        for (int tc = 0; tc < 4; ++tc) {
            int row = m0 + wr * 32 + tr * 16 + khalf * 8;
            int col = n0 + wc * 64 + tc * 16 + lrow;
            float bb = Bv[col];
#pragma unroll
            for (int r = 0; r < 8; ++r) {
                O[(size_t)(row + r) * Ndim + col] = acc[tr][tc][r] + bb;
            }
        }
    }
}

// ---------------------------------------------------------------------------
// Host-side orchestration
// ---------------------------------------------------------------------------
extern "C" void kernel_launch(void* const* d_in, const int* in_sizes, int n_in,
                              void* d_out, int out_size, void* d_ws, size_t ws_size,
                              hipStream_t stream) {
    const float* x  = (const float*)d_in[0];   // [32768,1024]
    const float* w  = (const float*)d_in[1];   // [1024,1024]
    const float* b  = (const float*)d_in[2];   // [1024]
    const float* u0 = (const float*)d_in[3];   // [1,1024]
    float* out = (float*)d_out;

    const int N = 1024;
    float* ws   = (float*)d_ws;
    float* A    = ws;                          // 4 MB: A = W^T W
    float* A2   = ws + (1u << 20);             // 4 MB: A2 = A*A
    float* ua   = ws + (2u << 20);
    float* ub   = ua + N;
    float* vraw = ub + N;
    float* yt   = vraw + N;
    float* scal = yt + N;                      // 256 scalar slots

    // init: zero atomic slots, ||u0||^2 -> scal[0]
    init_norm_kernel<<<1, 256, 0, stream>>>(u0, scal, N);

    // A = W^T W ; A2 = A^T A = A*A (A symmetric)
    mtm_kernel<<<dim3(16, 16), 256, 0, stream>>>(w, A, N);
    mtm_kernel<<<dim3(16, 16), 256, 0, stream>>>(A, A2, N);

    // u99 ∝ u0 · A^99  =  (u0·A) · (A2)^49, normalizing each step
    gemv_norm_kernel<<<N, 256, 0, stream>>>(A, u0, ua, &scal[0], &scal[1], N);
    float* cin = ua;
    float* cout = ub;
    for (int i = 0; i < 49; ++i) {
        gemv_norm_kernel<<<N, 256, 0, stream>>>(A2, cin, cout, &scal[1 + i], &scal[2 + i], N);
        float* tmp = cin; cin = cout; cout = tmp;
    }
    // cin = u99 (unnormalized), its ||.||^2 in scal[50]
    // ||u99hat W^T||^2 -> scal[60]   (v_raw = W * u99hat, row dots)
    gemv_norm_kernel<<<N, 256, 0, stream>>>(w, cin, vraw, &scal[50], &scal[60], N);
    // ||u99hat A||^2 -> scal[61]
    gemv_norm_kernel<<<N, 256, 0, stream>>>(A, cin, yt, &scal[50], &scal[61], N);

    // inv = 1 / max(1, sigma/0.9) -> scal[64]
    factor_kernel<<<1, 1, 0, stream>>>(scal);

    // out = x @ (W * inv) + b   (bf16x3 WMMA)
    gemm_bf16x3_kernel<<<dim3(8, 256), 256, 0, stream>>>(x, w, b, &scal[64], out,
                                                         32768, 1024, 1024);
}
